// AttMatch_2353642078852
// MI455X (gfx1250) — matrix operand order repaired and checked by reading, not compiled, not run on hardware
//
#include <hip/hip_runtime.h>
#include <hip/hip_bf16.h>
#include <math.h>

// ---------------------------------------------------------------------------
// Types for CDNA5 WMMA (wave32, 16x16x32 bf16 -> f32)
// ---------------------------------------------------------------------------
typedef __attribute__((ext_vector_type(16))) __bf16 v16bf;
typedef __attribute__((ext_vector_type(8)))  __bf16 v8bf;
typedef __attribute__((ext_vector_type(8)))  float  v8f;

// A-matrix fragment (16x32 bf16). Lane l: half h = l>>4, row m = l&15.
// Element j (0..7)  holds K = 8h + j          -> contiguous 8 @ base+8h
// Element j (8..15) holds K = 16 + 8h + (j-8) -> contiguous 8 @ base+16+8h
__device__ __forceinline__ v16bf frag_a(const __bf16* base, int h) {
  union { v16bf v; v8bf p[2]; } u;
  u.p[0] = *(const v8bf*)(base + 8 * h);
  u.p[1] = *(const v8bf*)(base + 16 + 8 * h);
  return u.v;
}

// B-matrix fragment (32x16 bf16), gathered from B^T row (n fixed, K contiguous).
// Lane l: n = l&15, h = l>>4; needs K = 8h..8h+7 and 16+8h..16+8h+7 (same as A).
__device__ __forceinline__ v16bf frag_b(const __bf16* base, int h) {
  union { v16bf v; v8bf p[2]; } u;
  u.p[0] = *(const v8bf*)(base + 8 * h);
  u.p[1] = *(const v8bf*)(base + 16 + 8 * h);
  return u.v;
}

__device__ __forceinline__ v8f wmma_bf16(v16bf a, v16bf b, v8f c) {
  // (neg_a, A, neg_b, B, c_mod, C, reuse_a, reuse_b)
  return __builtin_amdgcn_wmma_f32_16x16x32_bf16(false, a, false, b, (short)0, c,
                                                 false, false);
}

// ---------------------------------------------------------------------------
// Generic GEMM:  C[M,N] = A[M,K](bf16) * Bt[N,K]^T(bf16) + bias, with epilogs.
// Block tile 128x128, 8 waves (4 along M, 2 along N), wave tile 32x64.
// LDS tiles are double-buffered: stage K-step i+1 while WMMAs consume step i.
// ---------------------------------------------------------------------------
enum { EP_BIAS = 1, EP_RELU = 2, EP_SIG = 4, EP_F32 = 8, EP_BF16 = 16, EP_BF16T = 32 };

template <int EP>
__global__ __launch_bounds__(256) void gemm_bt_kernel(
    const __bf16* __restrict__ A, const __bf16* __restrict__ Bt,
    const float* __restrict__ bias, float* __restrict__ Cf,
    __bf16* __restrict__ Cb, __bf16* __restrict__ CbT, int M, int N, int K) {
  constexpr int AS = 40;  // padded LDS stride (bf16): 80B, bank-conflict free, 16B aligned
  __shared__ __bf16 As[2][128 * AS];
  __shared__ __bf16 Bs[2][128 * AS];
  const int tid = threadIdx.x;
  const int wave = tid >> 5, lane = tid & 31;
  const int wm = wave & 3, wn = wave >> 2;
  const int h = lane >> 4, ln = lane & 15;
  const int m0 = blockIdx.x * 128, n0 = blockIdx.y * 128;

  auto stage = [&](int buf, int k0) {
    for (int i = tid; i < 512; i += 256) {
      const int r = i >> 2, cc = (i & 3) << 3;
      *(v8bf*)&As[buf][r * AS + cc] = *(const v8bf*)&A[(size_t)(m0 + r) * K + k0 + cc];
      *(v8bf*)&Bs[buf][r * AS + cc] = *(const v8bf*)&Bt[(size_t)(n0 + r) * K + k0 + cc];
    }
  };

  v8f acc[2][4];
#pragma unroll
  for (int mi = 0; mi < 2; mi++)
#pragma unroll
    for (int ni = 0; ni < 4; ni++) acc[mi][ni] = (v8f)0.0f;

  stage(0, 0);
  int buf = 0;
  for (int k0 = 0; k0 < K; k0 += 32) {
    __syncthreads();
    if (k0 + 32 < K) stage(buf ^ 1, k0 + 32);  // prefetch next tile into other buffer
    v16bf af[2], bf[4];
#pragma unroll
    for (int mi = 0; mi < 2; mi++)
      af[mi] = frag_a(&As[buf][(wm * 32 + mi * 16 + ln) * AS], h);
#pragma unroll
    for (int ni = 0; ni < 4; ni++)
      bf[ni] = frag_b(&Bs[buf][(wn * 64 + ni * 16 + ln) * AS], h);
#pragma unroll
    for (int mi = 0; mi < 2; mi++)
#pragma unroll
      for (int ni = 0; ni < 4; ni++) acc[mi][ni] = wmma_bf16(af[mi], bf[ni], acc[mi][ni]);
    buf ^= 1;
  }

  // epilogue: C layout -> m = e + 8h (+tile), n = ln (+tile)
#pragma unroll
  for (int mi = 0; mi < 2; mi++) {
#pragma unroll
    for (int ni = 0; ni < 4; ni++) {
      const int n = n0 + wn * 64 + ni * 16 + ln;
      const float bvv = (EP & EP_BIAS) ? bias[n] : 0.0f;
      float vals[8];
#pragma unroll
      for (int e = 0; e < 8; e++) {
        float x = acc[mi][ni][e] + bvv;
        if (EP & EP_RELU) x = fmaxf(x, 0.0f);
        if (EP & EP_SIG) x = 1.0f / (1.0f + __expf(-x));
        vals[e] = x;
      }
      const int mb = m0 + wm * 32 + mi * 16 + 8 * h;
      if (EP & EP_F32) {
#pragma unroll
        for (int e = 0; e < 8; e++) Cf[(size_t)(mb + e) * N + n] = vals[e];
      }
      if (EP & EP_BF16) {
#pragma unroll
        for (int e = 0; e < 8; e++) Cb[(size_t)(mb + e) * N + n] = (__bf16)vals[e];
      }
      if (EP & EP_BF16T) {  // transposed store: 8 consecutive m -> one 16B store
        v8bf pk;
#pragma unroll
        for (int e = 0; e < 8; e++) pk[e] = (__bf16)vals[e];
        *(v8bf*)&CbT[(size_t)n * M + mb] = pk;
      }
    }
  }
}

// ---------------------------------------------------------------------------
// Flash attention: 128 queries/block, stream all T keys in 32-key chunks.
// O[m,:] = (1/l_m) * sum_t exp(s - m_run) * V[t,:]  (online softmax, fp32 state)
// Q fragments live in registers; scores go through LDS for the row reduction,
// exp-weights re-enter WMMA as bf16 (A operand of PV). All four B fragments of
// a chunk are loaded before any WMMA (pinned with a scheduling barrier) so the
// global loads pipeline and the latency is paid once per chunk.
// ---------------------------------------------------------------------------
__global__ __launch_bounds__(256) void flash_attn_kernel(
    const __bf16* __restrict__ Qg, const __bf16* __restrict__ Kg,
    const __bf16* __restrict__ VTg, float* __restrict__ Og, int T, float scale) {
  constexpr int SS = 36;  // score LDS stride (f32)
  constexpr int PS = 40;  // prob  LDS stride (bf16)
  __shared__ float  s_s[128 * SS];
  __shared__ __bf16 p_s[128 * PS];
  __shared__ float  m_run[128], l_run[128], corr[128];

  const int tid = threadIdx.x;
  const int wave = tid >> 5, lane = tid & 31;
  const int h = lane >> 4, ln = lane & 15;
  const int wm = wave & 3, wn = wave >> 2;
  const int m0 = blockIdx.x * 128;

  // q fragments: rows m0+wm*32+mi*16+ln, all K=128 (4 k-steps)
  v16bf aq[4][2];
#pragma unroll
  for (int kk = 0; kk < 4; kk++)
#pragma unroll
    for (int mi = 0; mi < 2; mi++)
      aq[kk][mi] = frag_a(&Qg[(size_t)(m0 + wm * 32 + mi * 16 + ln) * 128 + kk * 32], h);

  v8f om[2][4];
#pragma unroll
  for (int mi = 0; mi < 2; mi++)
#pragma unroll
    for (int di = 0; di < 4; di++) om[mi][di] = (v8f)0.0f;

  if (tid < 128) { m_run[tid] = -1e30f; l_run[tid] = 0.0f; corr[tid] = 1.0f; }
  __syncthreads();

  for (int c0 = 0; c0 < T; c0 += 32) {
    // prefetch next chunk's key rows / value columns (global_prefetch_b8)
    if (c0 + 32 < T) {
      __builtin_prefetch(&Kg[(size_t)(c0 + 32 + wn * 16 + ln) * 128], 0, 3);
      __builtin_prefetch(&VTg[(size_t)(wn * 64 + ln) * T + c0 + 32], 0, 3);
    }

    // ---- S = (Q K^T) * scale : tile 128 x 32 (waves: 4 along m, 2 along key)
    v16bf bk[4];
#pragma unroll
    for (int kk = 0; kk < 4; kk++)
      bk[kk] = frag_b(&Kg[(size_t)(c0 + wn * 16 + ln) * 128 + kk * 32], h);
    __builtin_amdgcn_sched_barrier(0);  // issue all loads before any WMMA
    v8f sc[2];
    sc[0] = (v8f)0.0f; sc[1] = (v8f)0.0f;
#pragma unroll
    for (int kk = 0; kk < 4; kk++) {
      sc[0] = wmma_bf16(aq[kk][0], bk[kk], sc[0]);
      sc[1] = wmma_bf16(aq[kk][1], bk[kk], sc[1]);
    }
#pragma unroll
    for (int mi = 0; mi < 2; mi++)
#pragma unroll
      for (int e = 0; e < 8; e++)
        s_s[(wm * 32 + mi * 16 + 8 * h + e) * SS + wn * 16 + ln] = sc[mi][e] * scale;
    __syncthreads();

    // ---- online softmax: 2 threads per row, 16 cols each
    {
      const int row = tid >> 1, half = tid & 1, cb = half * 16;
      float sv[16], mx = -1e30f;
#pragma unroll
      for (int c = 0; c < 16; c++) { sv[c] = s_s[row * SS + cb + c]; mx = fmaxf(mx, sv[c]); }
      mx = fmaxf(mx, __shfl_xor(mx, 1));
      const float mold = m_run[row];
      const float mnew = fmaxf(mold, mx);
      float sum = 0.0f;
#pragma unroll
      for (int c = 0; c < 16; c++) {
        const float p = __expf(sv[c] - mnew);
        sum += p;
        p_s[row * PS + cb + c] = (__bf16)p;
      }
      sum += __shfl_xor(sum, 1);
      if (half == 0) {
        const float cf = __expf(mold - mnew);
        l_run[row] = l_run[row] * cf + sum;
        m_run[row] = mnew;
        corr[row] = cf;
      }
    }
    __syncthreads();

    // ---- rescale O, then O += P V  (waves: 4 along m, 2 along dv)
    v16bf bv[4];
#pragma unroll
    for (int di = 0; di < 4; di++)
      bv[di] = frag_b(&VTg[(size_t)(wn * 64 + di * 16 + ln) * T + c0], h);
    __builtin_amdgcn_sched_barrier(0);  // pin V loads first; rescale VALU hides latency
#pragma unroll
    for (int mi = 0; mi < 2; mi++) {
      float cr[8];
#pragma unroll
      for (int e = 0; e < 8; e++) cr[e] = corr[wm * 32 + mi * 16 + 8 * h + e];
#pragma unroll
      for (int di = 0; di < 4; di++)
#pragma unroll
        for (int e = 0; e < 8; e++) om[mi][di][e] *= cr[e];
    }
    v16bf ap[2];
#pragma unroll
    for (int mi = 0; mi < 2; mi++) ap[mi] = frag_a(&p_s[(wm * 32 + mi * 16 + ln) * PS], h);
#pragma unroll
    for (int di = 0; di < 4; di++) {
      om[0][di] = wmma_bf16(ap[0], bv[di], om[0][di]);
      om[1][di] = wmma_bf16(ap[1], bv[di], om[1][di]);
    }
    __syncthreads();
  }

  // ---- finalize: O /= l  (double-normalization factor 1/(1+1e-16) == 1 in fp32)
#pragma unroll
  for (int mi = 0; mi < 2; mi++) {
    float il[8];
#pragma unroll
    for (int e = 0; e < 8; e++) il[e] = 1.0f / l_run[wm * 32 + mi * 16 + 8 * h + e];
#pragma unroll
    for (int di = 0; di < 4; di++) {
      const int dv = wn * 64 + di * 16 + ln;
#pragma unroll
      for (int e = 0; e < 8; e++)
        Og[(size_t)(m0 + wm * 32 + mi * 16 + 8 * h + e) * 128 + dv] = om[mi][di][e] * il[e];
    }
  }
}

// ---------------------------------------------------------------------------
// Elementwise / prep kernels
// ---------------------------------------------------------------------------
__global__ void cvt_bf16_kernel(const float* __restrict__ s, __bf16* __restrict__ d, int n) {
  const int i = blockIdx.x * blockDim.x + threadIdx.x;
  if (i < n) d[i] = (__bf16)s[i];
}

__global__ void concat_kernel(const float* __restrict__ a, const float* __restrict__ b,
                              float* __restrict__ o, int n) {
  const int i = blockIdx.x * blockDim.x + threadIdx.x;
  if (i < 2 * n) o[i] = (i < n) ? a[i] : b[i - n];
}

// W[K,N] f32 -> WT[N,K] bf16
__global__ void wtrans_kernel(const float* __restrict__ W, __bf16* __restrict__ WT, int K, int N) {
  const int i = blockIdx.x * blockDim.x + threadIdx.x;
  if (i >= K * N) return;
  const int k = i / N, n = i % N;
  WT[(size_t)n * K + k] = (__bf16)W[i];
}

// Combined SAGE weight: WlrT[n][k] = (k<256 ? Wl[k][n] : Wr[k-256][n]), bf16 [128,512]
__global__ void wlr_kernel(const float* __restrict__ Wl, const float* __restrict__ Wr,
                           __bf16* __restrict__ WT) {
  const int i = blockIdx.x * blockDim.x + threadIdx.x;
  if (i >= 128 * 512) return;
  const int n = i >> 9, k = i & 511;
  const float v = (k < 256) ? Wl[(size_t)k * 128 + n] : Wr[(size_t)(k - 256) * 128 + n];
  WT[(size_t)n * 512 + k] = (__bf16)v;
}

// feats = [X, X - out] -> f32 scratch + bf16 into A2 cols [256..511]
__global__ void feats_kernel(const float* __restrict__ X, const float* __restrict__ O,
                             float* __restrict__ featsF, __bf16* __restrict__ A2) {
  const int i = blockIdx.x * blockDim.x + threadIdx.x;
  if (i >= 8192 * 256) return;
  const int row = i >> 8, c = i & 255;
  float x;
  if (c < 128) x = X[(size_t)row * 128 + c];
  else { const int cc = c - 128; x = X[(size_t)row * 128 + cc] - O[(size_t)row * 128 + cc]; }
  featsF[(size_t)row * 256 + c] = x;
  A2[(size_t)row * 512 + 256 + c] = (__bf16)x;
}

__global__ void zero_kernel(float* __restrict__ p, int n) {
  const int i = blockIdx.x * blockDim.x + threadIdx.x;
  if (i < n) p[i] = 0.0f;
}

// edge scatter: agg[dst] += feats[src] (256 ch), deg[dst] += 1
__global__ void scatter_edges_kernel(const int* __restrict__ ei, const float* __restrict__ feats,
                                     float* __restrict__ agg, float* __restrict__ deg, int E) {
  const int i = blockIdx.x * blockDim.x + threadIdx.x;
  if (i >= E * 64) return;
  const int e = i >> 6, c0 = (i & 63) << 2;
  const int src = ei[e], dst = ei[E + e];
  const float4 v = *(const float4*)&feats[(size_t)src * 256 + c0];
  float* a = &agg[(size_t)dst * 256 + c0];
  atomicAdd(a + 0, v.x); atomicAdd(a + 1, v.y);
  atomicAdd(a + 2, v.z); atomicAdd(a + 3, v.w);
  if (c0 == 0) atomicAdd(&deg[dst], 1.0f);
}

// mean = agg / max(deg,1) -> bf16 into A2 cols [0..255]
__global__ void mean_kernel(const float* __restrict__ agg, const float* __restrict__ deg,
                            __bf16* __restrict__ A2) {
  const int i = blockIdx.x * blockDim.x + threadIdx.x;
  if (i >= 8192 * 256) return;
  const int row = i >> 8, c = i & 255;
  const float d = fmaxf(deg[row], 1.0f);
  A2[(size_t)row * 512 + c] = (__bf16)(agg[(size_t)row * 256 + c] / d);
}

// ---------------------------------------------------------------------------
// Host launcher
// ---------------------------------------------------------------------------
extern "C" void kernel_launch(void* const* d_in, const int* in_sizes, int n_in,
                              void* d_out, int out_size, void* d_ws, size_t ws_size,
                              hipStream_t stream) {
  (void)in_sizes; (void)n_in; (void)out_size; (void)ws_size;
  const float* x1 = (const float*)d_in[0];
  const float* x2 = (const float*)d_in[1];
  const int* ei[2] = { (const int*)d_in[2], (const int*)d_in[3] };
  const float* Wk[2] = { (const float*)d_in[4],  (const float*)d_in[13] };
  const float* bk[2] = { (const float*)d_in[5],  (const float*)d_in[14] };
  const float* Wq[2] = { (const float*)d_in[6],  (const float*)d_in[15] };
  const float* bq[2] = { (const float*)d_in[7],  (const float*)d_in[16] };
  const float* Wv[2] = { (const float*)d_in[8],  (const float*)d_in[17] };
  const float* bv[2] = { (const float*)d_in[9],  (const float*)d_in[18] };
  const float* Wl[2] = { (const float*)d_in[10], (const float*)d_in[19] };
  const float* bl[2] = { (const float*)d_in[11], (const float*)d_in[20] };
  const float* Wr[2] = { (const float*)d_in[12], (const float*)d_in[21] };

  char* ws = (char*)d_ws;
  size_t off = 0;
  auto alloc = [&](size_t bytes) -> void* {
    off = (off + 255) & ~(size_t)255;
    void* p = ws + off;
    off += bytes;
    return p;
  };

  __bf16 *WkT[2], *WqT[2], *WvT[2], *WlrT[2];
  for (int l = 0; l < 2; l++) {
    WkT[l] = (__bf16*)alloc(128 * 128 * 2);
    WqT[l] = (__bf16*)alloc(128 * 128 * 2);
    WvT[l] = (__bf16*)alloc(128 * 128 * 2);
    WlrT[l] = (__bf16*)alloc(128 * 512 * 2);
  }
  float*  Xcur  = (float*) alloc((size_t)8192 * 128 * 4);
  __bf16* Xb    = (__bf16*)alloc((size_t)8192 * 128 * 2);
  __bf16* Kb    = (__bf16*)alloc((size_t)8192 * 128 * 2);
  __bf16* Qb    = (__bf16*)alloc((size_t)8192 * 128 * 2);
  __bf16* VT    = (__bf16*)alloc((size_t)8192 * 128 * 2);  // [128][8192]
  float*  Oatt  = (float*) alloc((size_t)8192 * 128 * 4);
  float*  featsF= (float*) alloc((size_t)8192 * 256 * 4);
  __bf16* A2    = (__bf16*)alloc((size_t)8192 * 512 * 2);  // [mean | feats]
  float*  agg   = (float*) alloc((size_t)8192 * 256 * 4);
  float*  deg   = (float*) alloc((size_t)8192 * 4);
  __bf16* Fb    = (__bf16*)alloc((size_t)8192 * 128 * 2);

  const float scale = 0.088388347648318447f;  // 1/sqrt(128)
  const int E = 65536;

  // --- weight prep (bf16, transposed to [N][K]) ---
  for (int l = 0; l < 2; l++) {
    wtrans_kernel<<<64, 256, 0, stream>>>(Wk[l], WkT[l], 128, 128);
    wtrans_kernel<<<64, 256, 0, stream>>>(Wq[l], WqT[l], 128, 128);
    wtrans_kernel<<<64, 256, 0, stream>>>(Wv[l], WvT[l], 128, 128);
    wlr_kernel<<<256, 256, 0, stream>>>(Wl[l], Wr[l], WlrT[l]);
  }
  concat_kernel<<<4096, 256, 0, stream>>>(x1, x2, Xcur, 4096 * 128);

  for (int l = 0; l < 2; l++) {
    cvt_bf16_kernel<<<4096, 256, 0, stream>>>(Xcur, Xb, 8192 * 128);
    // K/Q/V projections (K=8192x128, bias, bf16 out; V also transposed for PV)
    gemm_bt_kernel<EP_BIAS | EP_BF16><<<dim3(64, 1), 256, 0, stream>>>(
        Xb, WkT[l], bk[l], nullptr, Kb, nullptr, 8192, 128, 128);
    gemm_bt_kernel<EP_BIAS | EP_BF16T><<<dim3(64, 1), 256, 0, stream>>>(
        Xb, WvT[l], bv[l], nullptr, nullptr, VT, 8192, 128, 128);
    gemm_bt_kernel<EP_BIAS | EP_BF16><<<dim3(64, 1), 256, 0, stream>>>(
        Xb, WqT[l], bq[l], nullptr, Qb, nullptr, 8192, 128, 128);
    // flash attention per graph: queries = graph rows, keys/values = all 8192 targets
    for (int g = 0; g < 2; g++)
      flash_attn_kernel<<<32, 256, 0, stream>>>(
          Qb + (size_t)g * 4096 * 128, Kb, VT, Oatt + (size_t)g * 4096 * 128, 8192, scale);
    // feats = [X, X - out]; mean aggregation over edges; SAGE as one K=512 GEMM
    feats_kernel<<<8192, 256, 0, stream>>>(Xcur, Oatt, featsF, A2);
    zero_kernel<<<8192, 256, 0, stream>>>(agg, 8192 * 256);
    zero_kernel<<<32, 256, 0, stream>>>(deg, 8192);
    for (int g = 0; g < 2; g++)
      scatter_edges_kernel<<<16384, 256, 0, stream>>>(
          ei[g], featsF + (size_t)g * 4096 * 256, agg + (size_t)g * 4096 * 256,
          deg + g * 4096, E);
    mean_kernel<<<8192, 256, 0, stream>>>(agg, deg, A2);
    if (l == 0)
      gemm_bt_kernel<EP_BIAS | EP_F32 | EP_RELU><<<dim3(64, 1), 256, 0, stream>>>(
          A2, WlrT[l], bl[l], Xcur, nullptr, nullptr, 8192, 128, 512);
    else
      gemm_bt_kernel<EP_BIAS | EP_F32><<<dim3(64, 1), 256, 0, stream>>>(
          A2, WlrT[l], bl[l], Xcur, nullptr, nullptr, 8192, 128, 512);
  }

  // --- adjacency: sigmoid(feats @ feats^T), 8192x8192 f32 to d_out ---
  cvt_bf16_kernel<<<4096, 256, 0, stream>>>(Xcur, Fb, 8192 * 128);
  gemm_bt_kernel<EP_SIG | EP_F32><<<dim3(64, 64), 256, 0, stream>>>(
      Fb, Fb, nullptr, (float*)d_out, nullptr, nullptr, 8192, 8192, 128);
}
